// CascadedMultiHeadAttention_53094385713870
// MI455X (gfx1250) — matrix operand order repaired and checked by reading, not compiled, not run on hardware
//
#include <hip/hip_runtime.h>
#include <hip/hip_bf16.h>
#include <cstdint>

typedef _Float16 half8 __attribute__((ext_vector_type(8)));
typedef _Float16 v16h  __attribute__((ext_vector_type(16)));
typedef float    v8f   __attribute__((ext_vector_type(8)));

#define BB 4
#define SS 2048
#define DD 640
#define HH 5
#define HD 128
#define K3 1920          // 3*DD (im2col K)
#define NT 3840          // 6*DD  (6 convs concatenated)
#define SP (SS + 2)      // padded sequence length
#define NTILES (NT / 16) // 240 column tiles

static constexpr size_t QKVE = (size_t)BB * HH * SS * HD;  // elems per tensor

__device__ __forceinline__ v8f wmma_f16(v16h a, v16h b, v8f c) {
  return __builtin_amdgcn_wmma_f32_16x16x32_f16(false, a, false, b, (short)0, c,
                                                false, false);
}

// ---------------------------------------------------------------------------
// Pack x (fp32 [B,S,D]) into zero-padded f16 [B, S+2, D].
// ---------------------------------------------------------------------------
__global__ void __launch_bounds__(256)
pack_x_f16(const float* __restrict__ x, _Float16* __restrict__ xh, int n) {
  int idx = blockIdx.x * 256 + threadIdx.x;
  if (idx >= n) return;
  int ci = idx % DD;
  int r  = (idx / DD) % SP;
  int b  = idx / (SP * DD);
  float v = (r == 0 || r == SP - 1) ? 0.f
                                    : x[((size_t)b * SS + (r - 1)) * DD + ci];
  xh[idx] = (_Float16)v;
}

// ---------------------------------------------------------------------------
// Pack 6 conv weights [D,D,3] fp32 directly into the gfx1250 WMMA B-operand
// register layout, tiled [ktile(60)][ntile(240)][lane(32)][e(16)] f16:
//   k = ktile*32 + (e<8 ? (lane>>4)*8+e : 16+(lane>>4)*8+e-8)   (B 32x16)
//   n = ntile*16 + (lane&15)
// The conv kernel then reads each B fragment as one contiguous 32B load.
// ---------------------------------------------------------------------------
__global__ void __launch_bounds__(256)
pack_w_f16(const float* w0, const float* w1, const float* w2,
           const float* w3, const float* w4, const float* w5,
           const float* b0, const float* b1, const float* b2,
           const float* b3, const float* b4, const float* b5,
           _Float16* __restrict__ whp, float* __restrict__ biascat, int n) {
  int idx = blockIdx.x * 256 + threadIdx.x;
  if (idx >= n) return;
  int e     = idx & 15;
  int lane  = (idx >> 4) & 31;
  int tile  = idx >> 9;
  int ntile = tile % NTILES;
  int ktile = tile / NTILES;
  int k  = ktile * 32 + (e < 8 ? (lane >> 4) * 8 + e : 16 + (lane >> 4) * 8 + (e - 8));
  int nn = ntile * 16 + (lane & 15);
  int t = k / DD, ci = k % DD;
  int c = nn / DD, co = nn % DD;
  const float* ws;
  switch (c) { case 0: ws = w0; break; case 1: ws = w1; break;
               case 2: ws = w2; break; case 3: ws = w3; break;
               case 4: ws = w4; break; default: ws = w5; }
  whp[idx] = (_Float16)ws[((size_t)co * DD + ci) * 3 + t];
  if (idx < NT) {
    int cb = idx / DD, cob = idx % DD;
    const float* bs;
    switch (cb) { case 0: bs = b0; break; case 1: bs = b1; break;
                  case 2: bs = b2; break; case 3: bs = b3; break;
                  case 4: bs = b4; break; default: bs = b5; }
    biascat[idx] = bs[cob];
  }
}

// ---------------------------------------------------------------------------
// Conv-as-GEMM, 2 M-tiles per wave sharing one B-stream:
// stage 34 overlapping x-rows (43.5KB) in LDS once, then a barrier-free
// K-loop: 4 ds_load_b128 (A0,A1) + 2 global b128 (B) + 2 WMMA per step.
// Epilogue: bias + ELU/tanh; Q/K -> [B,H,S,HD], V -> transposed [B,H,HD,S].
// ---------------------------------------------------------------------------
__global__ void __launch_bounds__(256)
conv_gemm_wmma(const _Float16* __restrict__ xh, const _Float16* __restrict__ whp,
               const float* __restrict__ biascat,
               _Float16* __restrict__ qk,    // 4 tensors: q1,k1,q2,k2
               _Float16* __restrict__ vt1,   // [B,H,HD,S]
               _Float16* __restrict__ vt2) { // [B,H,HD,S]
  __shared__ __align__(16) _Float16 Ax[34 * DD];
  const int tid = threadIdx.x;
  const int wave = tid >> 5, lane = tid & 31, hf = lane >> 4, m = lane & 15;
  const int n0 = blockIdx.x * 128;
  const int s0 = blockIdx.y * 32;
  const int bz = blockIdx.z;
  const size_t xbase = ((size_t)bz * SP + s0) * DD;

  // Stage 34 contiguous rows of padded x (im2col panel for 32 outputs).
  for (int i = tid; i < (34 * DD) / 8; i += 256)
    *(uint4*)&Ax[i * 8] = *(const uint4*)(xh + xbase + (size_t)i * 8);
  __syncthreads();

  const int ntile = blockIdx.x * 8 + wave;
  const size_t bstride = (size_t)NTILES * 512;  // halves per ktile
  const _Float16* bp = whp + ((size_t)ntile * 32 + lane) * 16;

  const v8f vzero = {0.f, 0.f, 0.f, 0.f, 0.f, 0.f, 0.f, 0.f};
  v8f acc0 = vzero, acc1 = vzero;
  #pragma unroll
  for (int t = 0; t < 3; ++t) {
    const _Float16* arow0 = &Ax[(m + t) * DD];
    const _Float16* arow1 = &Ax[(16 + m + t) * DD];
    #pragma unroll 4
    for (int c20 = 0; c20 < 20; ++c20) {
      const _Float16* ar0 = arow0 + c20 * 32;
      half8 a0lo = *(const half8*)(ar0 + hf * 8);
      half8 a0hi = *(const half8*)(ar0 + 16 + hf * 8);
      v16h a0 = __builtin_shufflevector(a0lo, a0hi, 0, 1, 2, 3, 4, 5, 6, 7,
                                        8, 9, 10, 11, 12, 13, 14, 15);
      const _Float16* ar1 = arow1 + c20 * 32;
      half8 a1lo = *(const half8*)(ar1 + hf * 8);
      half8 a1hi = *(const half8*)(ar1 + 16 + hf * 8);
      v16h a1 = __builtin_shufflevector(a1lo, a1hi, 0, 1, 2, 3, 4, 5, 6, 7,
                                        8, 9, 10, 11, 12, 13, 14, 15);
      v16h b = *(const v16h*)bp;
      __builtin_prefetch(bp + 2 * bstride, 0, 3);
      bp += bstride;
      acc0 = wmma_f16(a0, b, acc0);
      acc1 = wmma_f16(a1, b, acc1);
    }
  }

  // Epilogue: bias + activation, scatter (two 16-row groups).
  const int n_abs = n0 + wave * 16 + m;
  const int cidx = n_abs / DD, co = n_abs % DD;
  const int h = co >> 7, d = co & 127;
  const float bias = biascat[n_abs];
  #pragma unroll
  for (int g = 0; g < 2; ++g) {
    const v8f acc = g ? acc1 : acc0;
    const int sbase = s0 + g * 16;
    if (cidx == 2 || cidx == 5) {            // V1 / V2 -> transposed layout
      _Float16* vt = (cidx == 2) ? vt1 : vt2;
      half8 hv;
      #pragma unroll
      for (int r = 0; r < 8; ++r) {
        float v = acc[r] + bias;
        v = (cidx == 5) ? tanhf(v) : (v > 0.f ? v : (__expf(v) - 1.f));
        hv[r] = (_Float16)v;
      }
      *(half8*)(vt + ((size_t)(bz * HH + h) * HD + d) * SS + sbase + hf * 8) = hv;
    } else {                                  // Q1,K1,Q2,K2 -> [B,H,S,HD]
      const int slot = (cidx == 0) ? 0 : (cidx == 1) ? 1 : (cidx == 3) ? 2 : 3;
      _Float16* dst = qk + (size_t)slot * QKVE +
                      (((size_t)bz * HH + h) * SS) * HD + d;
      #pragma unroll
      for (int r = 0; r < 8; ++r) {
        float v = acc[r] + bias;
        v = (v > 0.f) ? v : (__expf(v) - 1.f);
        dst[(size_t)(sbase + hf * 8 + r) * HD] = (_Float16)v;
      }
    }
  }
}

// ---------------------------------------------------------------------------
// Flash attention, one wave per 16 queries x HD=128, keys streamed in blocks
// of 32. Scores computed as S^T = K*Q^T so the WMMA C-layout (lane = query)
// equals the f16 A-operand layout for P*V: zero cross-lane movement.
// V is pre-transposed [B,H,HD,S] so each V fragment is two contiguous 16B loads along S.
// ---------------------------------------------------------------------------
__global__ void __launch_bounds__(256)
flash_attn_wmma(const _Float16* __restrict__ Q, const _Float16* __restrict__ K,
                const _Float16* __restrict__ VT, float* __restrict__ O) {
  const int w = (int)((blockIdx.x * blockDim.x + threadIdx.x) >> 5);
  if (w >= BB * HH * (SS / 16)) return;
  const int lane = threadIdx.x & 31, hf = lane >> 4, m = lane & 15;
  const int bh = w >> 7;
  const int q0 = (w & 127) << 4;
  const _Float16* Qb  = Q  + (size_t)bh * SS * HD;
  const _Float16* Kb  = K  + (size_t)bh * SS * HD;
  const _Float16* VTb = VT + (size_t)bh * HD * SS;

  v16h qf[4];
  #pragma unroll
  for (int ch = 0; ch < 4; ++ch)
    qf[ch] = *(const v16h*)(Qb + (size_t)(q0 + m) * HD + ch * 32 + hf * 16);

  v8f o[8];
  const v8f vzero = {0.f, 0.f, 0.f, 0.f, 0.f, 0.f, 0.f, 0.f};
  #pragma unroll
  for (int t = 0; t < 8; ++t) o[t] = vzero;
  float mr = -3.4e38f, lr = 0.f;
  const float scale = 0.08838834764831845f;  // 1/sqrt(128)

  for (int kb = 0; kb < SS / 32; ++kb) {
    const int key0 = kb * 32;
    __builtin_prefetch(Kb + (size_t)(key0 + 32 + m) * HD, 0, 3);
    v8f s0 = vzero, s1 = vzero;
    #pragma unroll
    for (int ch = 0; ch < 4; ++ch) {
      const _Float16* kp0 = Kb + (size_t)(key0 + m) * HD + ch * 32;
      half8 l0 = *(const half8*)(kp0 + hf * 16);
      half8 h0 = *(const half8*)(kp0 + hf * 16 + 8);
      v16h a0 = __builtin_shufflevector(l0, h0, 0, 1, 2, 3, 4, 5, 6, 7,
                                        8, 9, 10, 11, 12, 13, 14, 15);
      s0 = wmma_f16(a0, qf[ch], s0);
      const _Float16* kp1 = Kb + (size_t)(key0 + 16 + m) * HD + ch * 32;
      half8 l1 = *(const half8*)(kp1 + hf * 16);
      half8 h1 = *(const half8*)(kp1 + hf * 16 + 8);
      v16h a1 = __builtin_shufflevector(l1, h1, 0, 1, 2, 3, 4, 5, 6, 7,
                                        8, 9, 10, 11, 12, 13, 14, 15);
      s1 = wmma_f16(a1, qf[ch], s1);
    }
    // Online softmax; lane q holds keys {0..7,16..23}, lane 16+q the rest.
    float p0[8], p1[8];
    float rmax = -3.4e38f;
    #pragma unroll
    for (int r = 0; r < 8; ++r) {
      p0[r] = s0[r] * scale;
      p1[r] = s1[r] * scale;
      rmax = fmaxf(rmax, fmaxf(p0[r], p1[r]));
    }
    rmax = fmaxf(rmax, __shfl_xor(rmax, 16, 32));
    float mnew  = fmaxf(mr, rmax);
    float alpha = __expf(mr - mnew);
    float ls = 0.f;
    #pragma unroll
    for (int r = 0; r < 8; ++r) {
      p0[r] = __expf(p0[r] - mnew);
      p1[r] = __expf(p1[r] - mnew);
      ls += p0[r] + p1[r];
    }
    lr = lr * alpha + ls + __shfl_xor(ls, 16, 32);
    mr = mnew;
    v16h pa;
    #pragma unroll
    for (int r = 0; r < 8; ++r) {
      pa[r]     = (_Float16)p0[r];
      pa[8 + r] = (_Float16)p1[r];
    }
    float a8[8];
    #pragma unroll
    for (int r = 0; r < 8; ++r) a8[r] = __shfl(alpha, hf * 8 + r, 32);
    #pragma unroll
    for (int t = 0; t < 8; ++t) {
      #pragma unroll
      for (int r = 0; r < 8; ++r) o[t][r] *= a8[r];
    }
    // O += P * V, V fragment = two 16B loads (keys 8h..8h+7, 16+8h..) from transposed V.
    #pragma unroll
    for (int t = 0; t < 8; ++t) {
      const _Float16* vp = VTb + (size_t)(t * 16 + m) * SS + key0;
      half8 vlo = *(const half8*)(vp + hf * 8);
      half8 vhi = *(const half8*)(vp + 16 + hf * 8);
      v16h vb = __builtin_shufflevector(vlo, vhi, 0, 1, 2, 3, 4, 5, 6, 7,
                                        8, 9, 10, 11, 12, 13, 14, 15);
      o[t] = wmma_f16(pa, vb, o[t]);
    }
  }
  float linv[8];
  #pragma unroll
  for (int r = 0; r < 8; ++r) linv[r] = 1.f / __shfl(lr, hf * 8 + r, 32);
  float* Ob = O + (size_t)bh * SS * HD;
  #pragma unroll
  for (int t = 0; t < 8; ++t) {
    #pragma unroll
    for (int r = 0; r < 8; ++r)
      Ob[(size_t)(q0 + hf * 8 + r) * HD + t * 16 + m] = o[t][r] * linv[r];
  }
}

// ---------------------------------------------------------------------------
// out = LayerNorm(O1 * O2) over D=640, eps=1e-5. One block per (b,s).
// ---------------------------------------------------------------------------
__global__ void __launch_bounds__(256)
mul_layernorm(const float* __restrict__ O1, const float* __restrict__ O2,
              const float* __restrict__ g, const float* __restrict__ bt,
              float* __restrict__ out) {
  const int row = blockIdx.x;
  const int b = row >> 11, s = row & (SS - 1);
  const int tid = threadIdx.x;
  float vals[3];
  float sum = 0.f, sq = 0.f;
  int j = 0;
  for (int i = tid; i < DD; i += 256, ++j) {
    int h = i >> 7, d = i & 127;
    size_t off = (((size_t)b * HH + h) * SS + s) * HD + d;
    float v = O1[off] * O2[off];
    vals[j] = v;
    sum += v;
    sq += v * v;
  }
  __shared__ float rs[256], rq[256];
  rs[tid] = sum;
  rq[tid] = sq;
  __syncthreads();
  for (int o2 = 128; o2 > 0; o2 >>= 1) {
    if (tid < o2) { rs[tid] += rs[tid + o2]; rq[tid] += rq[tid + o2]; }
    __syncthreads();
  }
  const float mean = rs[0] * (1.f / DD);
  const float var  = rq[0] * (1.f / DD) - mean * mean;
  const float rstd = rsqrtf(var + 1e-5f);
  j = 0;
  for (int i = tid; i < DD; i += 256, ++j)
    out[(size_t)row * DD + i] = (vals[j] - mean) * rstd * g[i] + bt[i];
}

// ---------------------------------------------------------------------------
extern "C" void kernel_launch(void* const* d_in, const int* in_sizes, int n_in,
                              void* d_out, int out_size, void* d_ws,
                              size_t ws_size, hipStream_t stream) {
  (void)in_sizes; (void)n_in; (void)out_size; (void)ws_size;
  const float* x = (const float*)d_in[0];
  const float* w[6]  = {(const float*)d_in[1], (const float*)d_in[3],
                        (const float*)d_in[5], (const float*)d_in[7],
                        (const float*)d_in[9], (const float*)d_in[11]};
  const float* bi[6] = {(const float*)d_in[2], (const float*)d_in[4],
                        (const float*)d_in[6], (const float*)d_in[8],
                        (const float*)d_in[10], (const float*)d_in[12]};
  const float* ln_g = (const float*)d_in[13];
  const float* ln_b = (const float*)d_in[14];
  float* out = (float*)d_out;

  uint8_t* ws = (uint8_t*)d_ws;
  size_t off = 0;
  auto take = [&](size_t bytes) -> void* {
    void* p = ws + off;
    off = (off + bytes + 255) & ~(size_t)255;
    return p;
  };
  _Float16* xh      = (_Float16*)take((size_t)BB * SP * DD * 2);
  _Float16* whp     = (_Float16*)take((size_t)K3 * NT * 2);
  float*    biascat = (float*)take((size_t)NT * 4);
  _Float16* qk      = (_Float16*)take(4 * QKVE * 2);  // q1,k1,q2,k2
  _Float16* vt1     = (_Float16*)take(QKVE * 2);      // v1 transposed
  _Float16* vt2     = (_Float16*)take(QKVE * 2);      // v2 transposed
  float*    O1      = (float*)take(QKVE * 4);
  float*    O2      = (float*)take(QKVE * 4);

  const int nx = BB * SP * DD;
  pack_x_f16<<<(nx + 255) / 256, 256, 0, stream>>>(x, xh, nx);
  const int nw = K3 * NT;
  pack_w_f16<<<(nw + 255) / 256, 256, 0, stream>>>(
      w[0], w[1], w[2], w[3], w[4], w[5],
      bi[0], bi[1], bi[2], bi[3], bi[4], bi[5], whp, biascat, nw);
  conv_gemm_wmma<<<dim3(NT / 128, SS / 32, BB), 256, 0, stream>>>(
      xh, whp, biascat, qk, vt1, vt2);
  const int attn_blocks = (BB * HH * (SS / 16)) / 8;
  flash_attn_wmma<<<attn_blocks, 256, 0, stream>>>(qk + 0 * QKVE,
                                                   qk + 1 * QKVE, vt1, O1);
  flash_attn_wmma<<<attn_blocks, 256, 0, stream>>>(qk + 2 * QKVE,
                                                   qk + 3 * QKVE, vt2, O2);
  mul_layernorm<<<BB * SS, 256, 0, stream>>>(O1, O2, ln_g, ln_b, out);
}
